// PoseHead_46110768890230
// MI455X (gfx1250) — matrix operand order repaired and checked by reading, not compiled, not run on hardware
//
#include <hip/hip_runtime.h>
#include <hip/hip_bf16.h>

// ---------------------------------------------------------------------------
// MI455X (gfx1250) fused keyframe-chunk transformer.
//   - bf16 WMMA (v_wmma_f32_16x16x32_bf16), wave32, 256-thread (8-wave) blocks
//   - LayerNorm FOLDED into pre-converted transposed bf16 weights:
//       h@W = r*(x@(g.W)) - mu*r*Sg + C   (Sg, C precomputed per column)
//   - chunk activation tile pulled into LDS by the Tensor Data Mover
//     (tensor_load_to_lds + s_wait_tensorcnt), bf16 feature staged in ws
//   - whole 64x768 chunk sequence kept in LDS (~305KB of the 320KB/WGP)
//   - pose head is a WMMA GEMM too; dead global-attention branch eliminated
// Input indexing assumes jax pytree (alphabetical-dict) flattening:
//   0:feature | 1..6:global_head | 7..54:global_params | 55..60:local_head |
//   61..108:local_params (per layer: W1,W2,Wo,Wqkv,b1,b2,bo,bqkv,
//                          ln1_b,ln1_g,ln2_b,ln2_g)
// ---------------------------------------------------------------------------

typedef __bf16 bf16;
typedef __attribute__((ext_vector_type(16))) __bf16 v16bf;
typedef __attribute__((ext_vector_type(8)))  __bf16 v8bf;
typedef __attribute__((ext_vector_type(8)))  float  v8f;
typedef __attribute__((ext_vector_type(4)))  unsigned int v4u;
typedef __attribute__((ext_vector_type(8)))  int v8i;
typedef __attribute__((ext_vector_type(4)))  int v4i;

#define BLK   256
#define WAVES 8

// ---- workspace: bf16 weights (half-element offsets) -----------------------
constexpr size_t W_QKVT = 0;              // [2304][768]  (g1 folded)
constexpr size_t W_WOT  = 1769472;        // [768][768]
constexpr size_t W_W1T  = 2359296;        // [3072][768]  (g2 folded)
constexpr size_t W_W2T  = 4718592;        // [768][3072]
constexpr size_t LSTR   = 7077888;        // halfs per layer
constexpr size_t W_HW1T = 4 * LSTR;       // head W1 folded+transposed [128][768]
// ---- f32 aux (C / Sg per folded matrix), int metadata, bf16 feature -------
constexpr size_t AUX_BYTE  = (W_HW1T + 98304) * 2;
constexpr size_t AUX_LAYER = 10752;       // floats: CQ 2304 SGQ 2304 C1 3072 SG1 3072
constexpr size_t AUX_HEAD  = 4 * AUX_LAYER;  // float offset of CH[128],SGH[128]
constexpr size_t META_BYTE = AUX_BYTE + (AUX_HEAD + 256) * 4;
constexpr size_t FEAT_BYTE = META_BYTE + 8192;   // bf16 feature [8*64*768]
constexpr int    FEAT_N    = 8 * 64 * 768;

// ---- LDS byte offsets (single dynamic allocation, phases overlap) ---------
constexpr int OFF_X   = 0;        // bf16 [64][768]   x (residual stream)
constexpr int OFF_RS  = 98304;    // f32  [64][2]     per-row mean/rstd
constexpr int OFF_SC  = 98816;    // f32  [64][64]    raw scores
constexpr int OFF_AT  = 115200;   // bf16 [64][64]    softmaxed probs
constexpr int OFF_Q   = 123392;   // bf16 [64][192]
constexpr int OFF_K   = 147968;   // bf16 [64][192]
constexpr int OFF_VT  = 172544;   // bf16 [192][64]   V stored transposed
constexpr int OFF_OH  = 197120;   // bf16 [64][192]   per-head attn output
constexpr int OFF_ACC = 98816;    // f32  [64][768]   FFN accumulator (phase 2)
constexpr int OFF_IB  = 295424;   // bf16 [64][128]   FFN block / pose hidden
constexpr int SMEM_C  = 311808;   // total dynamic LDS bytes (<= 320KB/WGP)

// ---------------------------------------------------------------------------
__device__ __forceinline__ v8f wmma_bf16(v16bf a, v16bf b, v8f c) {
  return __builtin_amdgcn_wmma_f32_16x16x32_bf16(false, a, false, b,
                                                 (short)0, c, false, false);
}

// A-fragment (16x32 bf16): lane L row M=L%16,
// element s -> K = kblk + (L>=16?16:0) + s (matches bfrag),
// i.e. one contiguous 16-half run -> two 16B LDS vector loads.
__device__ __forceinline__ v16bf afrag(const bf16* A, int pitch, int m0,
                                       int kblk, int lane) {
  const int hb = (lane >> 4) ? 16 : 0;
  const bf16* p = A + (size_t)(m0 + (lane & 15)) * pitch + kblk + hb;
  v8bf lo = *(const v8bf*)p;
  v8bf hi = *(const v8bf*)(p + 8);
  return __builtin_shufflevector(lo, hi, 0, 1, 2, 3, 4, 5, 6, 7,
                                 8, 9, 10, 11, 12, 13, 14, 15);
}

// B-fragment (32x16 bf16): lane L col N=L%16 gets 16 contiguous K values
// starting at kblk + (L>=16 ? 16 : 0) -> one aligned 32-byte vector load.
__device__ __forceinline__ v16bf bfrag(const bf16* Wrow, int kblk, int lane) {
  return *(const v16bf*)(Wrow + kblk + ((lane >> 4) ? 16 : 0));
}

// Per-row mean / rstd: 4 lanes per row, butterfly-combined with __shfl_xor.
__device__ __forceinline__ void row_stats(const bf16* X, float* RS, int tid) {
  const int row = tid >> 2, sub = tid & 3;
  const bf16* p = X + (size_t)row * 768 + sub * 192;
  float s = 0.f, q = 0.f;
  for (int k = 0; k < 192; ++k) { float x = (float)p[k]; s += x; q += x * x; }
  s += __shfl_xor(s, 1, 32); q += __shfl_xor(q, 1, 32);
  s += __shfl_xor(s, 2, 32); q += __shfl_xor(q, 2, 32);
  if (sub == 0) {
    float mu = s * (1.f / 768.f);
    float var = q * (1.f / 768.f) - mu * mu;
    RS[2 * row] = mu;
    RS[2 * row + 1] = rsqrtf(fmaxf(var, 0.f) + 1e-5f);
  }
}

// Q/K/V^T projection GEMM; MODE is compile-time so stores are straight-line
// (no exec-mask divergence).  MODE: 0=Q row-major, 1=K row-major, 2=V^T.
template <int MODE>
__device__ __forceinline__ void qkv_gemm(const bf16* X, const bf16* WqkvT,
                                         const float* CQ, const float* SGQ,
                                         const float* RS, bf16* dst, int hd,
                                         int wv, int lane, int hiRow) {
  for (int t = wv; t < 48; t += WAVES) {
    int mt = t / 12, nt = t % 12;
    int n = nt * 16 + (lane & 15);
    int gn = MODE * 768 + hd * 192 + n;
    v8f acc = {};
#pragma unroll 1
    for (int kb = 0; kb < 768; kb += 32) {
      v16bf a  = afrag(X, 768, mt * 16, kb, lane);
      v16bf bb = bfrag(WqkvT + (size_t)gn * 768, kb, lane);
      acc = wmma_bf16(a, bb, acc);
    }
    float cqv = CQ[gn], sgv = SGQ[gn];
#pragma unroll
    for (int v = 0; v < 8; ++v) {
      int m = mt * 16 + v + hiRow;
      float mu = RS[2 * m], rs = RS[2 * m + 1];
      float val = rs * acc[v] + cqv - mu * rs * sgv;
      if (MODE == 2) dst[n * 64 + m] = (bf16)val;      // transpose store
      else           dst[m * 192 + n] = (bf16)val;
    }
  }
}

// ---------------------------------------------------------------------------
// Kernel A1: plain f32 -> bf16 convert + transpose (Wo, W2)
__global__ __launch_bounds__(BLK)
void convert_transpose(const float* __restrict__ src, bf16* __restrict__ dst,
                       int K, int N) {
  size_t total = (size_t)K * N;
  for (size_t i = (size_t)blockIdx.x * BLK + threadIdx.x; i < total;
       i += (size_t)gridDim.x * BLK) {
    size_t k = i / N, n = i - k * N;
    dst[n * (size_t)K + k] = (bf16)src[i];
  }
}

// Kernel A2: fold LayerNorm into a weight matrix consumed by LN(x)@W:
//   WT[n][k] = g[k]*W[k][n] (bf16),  Sg[n] = sum_k g[k]W[k][n],
//   C[n]     = bias[n] + sum_k beta[k]*W[k][n]
__global__ __launch_bounds__(BLK)
void fold_ln_transpose(const float* __restrict__ W, const float* __restrict__ g,
                       const float* __restrict__ beta,
                       const float* __restrict__ bias, int K, int N,
                       bf16* __restrict__ WT, float* __restrict__ C,
                       float* __restrict__ Sg) {
  int n = blockIdx.x * BLK + threadIdx.x;
  if (n >= N) return;
  float c = bias[n], s = 0.f;
  for (int k = 0; k < K; ++k) {
    float w = W[(size_t)k * N + n];
    float gw = g[k] * w;
    WT[(size_t)n * K + k] = (bf16)gw;
    s += gw;
    c += beta[k] * w;
  }
  C[n] = c;
  Sg[n] = s;
}

// Kernel A3: elementwise f32 -> bf16 (stages feature for the TDM path)
__global__ __launch_bounds__(BLK)
void convert_cast(const float* __restrict__ src, bf16* __restrict__ dst, int N) {
  int i = blockIdx.x * BLK + threadIdx.x;
  if (i < N) dst[i] = (bf16)src[i];
}

// ---------------------------------------------------------------------------
// Kernel B: cosine-similarity keyframe scan -> chunk metadata (one wave/batch)
__global__ __launch_bounds__(BLK)
void scan_chunks(const float* __restrict__ feature, int* __restrict__ meta) {
  __shared__ int flag[8][64];
  const int wv = threadIdx.x >> 5, lane = threadIdx.x & 31;
  const float* f = feature + (size_t)wv * 64 * 768;
  float ref[24], cur[24];
  float ss = 0.f;
#pragma unroll
  for (int i = 0; i < 24; ++i) { cur[i] = f[i * 32 + lane]; ss += cur[i] * cur[i]; }
  for (int m = 16; m; m >>= 1) ss += __shfl_xor(ss, m, 32);
  float inv = 1.f / fmaxf(sqrtf(ss), 1e-12f);
#pragma unroll
  for (int i = 0; i < 24; ++i) ref[i] = cur[i] * inv;
  if (lane == 0) flag[wv][0] = 1;
  for (int t = 1; t < 64; ++t) {
    float s2 = 0.f, dt = 0.f;
#pragma unroll
    for (int i = 0; i < 24; ++i) {
      float x = f[(size_t)t * 768 + i * 32 + lane];
      cur[i] = x; s2 += x * x; dt += ref[i] * x;
    }
    for (int m = 16; m; m >>= 1) {
      s2 += __shfl_xor(s2, m, 32);
      dt += __shfl_xor(dt, m, 32);
    }
    float inv2 = 1.f / fmaxf(sqrtf(s2), 1e-12f);
    int key = (dt * inv2) < 0.95f;
    if (key) {
#pragma unroll
      for (int i = 0; i < 24; ++i) ref[i] = cur[i] * inv2;
    }
    if (lane == 0) flag[wv][t] = key;
  }
  if (lane == 0) {
    int* numc = meta; int* st = meta + 8; int* ln = meta + 8 + 512;
    int c = 0, prev = 0;
    st[wv * 64 + 0] = 0;
    for (int t = 1; t < 64; ++t) {
      if (flag[wv][t]) { ln[wv * 64 + c] = t - prev; ++c; st[wv * 64 + c] = t; prev = t; }
    }
    ln[wv * 64 + c] = 64 - prev;
    numc[wv] = c + 1;
  }
}

// ---------------------------------------------------------------------------
struct KParams {
  const bf16*  featbf;                       // bf16-staged feature
  const float* bo[4]; const float* b2[4];
  const float* cq[4]; const float* sgq[4];   // folded-LN scalars for Wqkv
  const float* c1[4]; const float* sg1[4];   // folded-LN scalars for W1
  const float* ch;    const float* sgh;      // folded-LN scalars for head W1
  const float* hb2;   const float* hW2;
  const bf16* wb; const int* meta; float* out;
};

extern __shared__ char smem[];

// Kernel C: one workgroup per (batch, chunk). 4-layer pre-LN transformer on a
// 64-slot sequence entirely in LDS, fused WMMA pose-head epilogue.
__global__ __launch_bounds__(BLK)
void chunk_transformer_pose(KParams P) {
  const int c = blockIdx.x, b = blockIdx.y;
  const int tid = threadIdx.x, lane = tid & 31, wv = tid >> 5;
  const int* numc   = P.meta;
  const int* startA = P.meta + 8;
  const int* lenA   = P.meta + 8 + 512;
  if (c >= numc[b]) return;                // block-uniform: safe before barriers
  const int start = startA[b * 64 + c], len = lenA[b * 64 + c];

  bf16*  X    = (bf16*)(smem + OFF_X);
  float* RS   = (float*)(smem + OFF_RS);
  float* SC   = (float*)(smem + OFF_SC);
  bf16*  AT   = (bf16*)(smem + OFF_AT);
  bf16*  Q    = (bf16*)(smem + OFF_Q);
  bf16*  Km   = (bf16*)(smem + OFF_K);
  bf16*  VT   = (bf16*)(smem + OFF_VT);
  bf16*  OH   = (bf16*)(smem + OFF_OH);
  float* ACC  = (float*)(smem + OFF_ACC);
  bf16*  IB   = (bf16*)(smem + OFF_IB);
  bf16*  HIDB = (bf16*)(smem + OFF_IB);    // pose hidden reuses IB region

  // ---- gather the chunk's frames into LDS --------------------------------
  const bf16* src = P.featbf + ((size_t)(b * 64) + start) * 768;
#if __has_builtin(__builtin_amdgcn_tensor_load_to_lds)
  // Tensor Data Mover: rows [start, start+len) are one contiguous byte range
  // (len*1536 B) -> single 1-D D# (data_size=8B, tile_dim0=len*192, type=2).
  if (wv == 0) {
    const unsigned td   = (unsigned)(len * 192);       // 8-byte elements
    const unsigned ldsa = (unsigned)OFF_X;             // LDS byte address of X
    const unsigned long long ga = (unsigned long long)(size_t)(const void*)src;
    v4u g0 = { 1u,                                     // count=1 (valid D#)
               ldsa,
               (unsigned)(ga & 0xffffffffu),
               (unsigned)((ga >> 32) & 0x1ffffffu) | 0x80000000u };  // type=2
    v8i g1 = { (int)0x30000,                           // data_size=3 (8B)
               (int)((td & 0xffffu) << 16),            // tensor_dim0 lo
               (int)(((td >> 16) & 0xffffu) | (1u << 16)),  // dim0 hi | dim1=1
               (int)((td & 0xffffu) << 16),            // tile_dim0
               1,                                      // tile_dim1=1
               (int)td,                                // tensor_dim0_stride
               0, 0 };
    v4i g2 = { 0, 0, 0, 0 }, g3 = { 0, 0, 0, 0 };
    v8i g4 = { 0, 0, 0, 0, 0, 0, 0, 0 };               // 6-arg toolchain form
    __builtin_amdgcn_tensor_load_to_lds(g0, g1, g2, g3, g4, 0);
#if __has_builtin(__builtin_amdgcn_s_wait_tensorcnt)
    __builtin_amdgcn_s_wait_tensorcnt(0);
#else
    asm volatile("s_wait_tensorcnt 0x0" ::: "memory");
#endif
  }
  for (int i = tid; i < (64 - len) * 768; i += BLK)    // zero the padded rows
    X[len * 768 + i] = (bf16)0.f;
#else
  for (int i = tid; i < 64 * 768; i += BLK)
    X[i] = (i < len * 768) ? src[i] : (bf16)0.f;
#endif
  __syncthreads();

  const float rscale = rsqrtf(192.f);
  const int hiRow = (lane >> 4) ? 8 : 0;

  for (int l = 0; l < 4; ++l) {
    const bf16* WqkvT = P.wb + (size_t)l * LSTR + W_QKVT;
    const bf16* WoT   = P.wb + (size_t)l * LSTR + W_WOT;
    const bf16* W1T   = P.wb + (size_t)l * LSTR + W_W1T;
    const bf16* W2T   = P.wb + (size_t)l * LSTR + W_W2T;
    if (tid == 0 && l + 1 < 4)           // warm next layer into L2/WGP$
      __builtin_prefetch(P.wb + (size_t)(l + 1) * LSTR, 0, 1);

    row_stats(X, RS, tid);               // ln1 stats (mu, rstd per row)
    __syncthreads();

    for (int hd = 0; hd < 4; ++hd) {
      // ---- Q/K/V projections; LN folded into WqkvT; uniform store paths ---
      qkv_gemm<0>(X, WqkvT, P.cq[l], P.sgq[l], RS, Q,  hd, wv, lane, hiRow);
      qkv_gemm<1>(X, WqkvT, P.cq[l], P.sgq[l], RS, Km, hd, wv, lane, hiRow);
      qkv_gemm<2>(X, WqkvT, P.cq[l], P.sgq[l], RS, VT, hd, wv, lane, hiRow);
      __syncthreads();
      // ---- scores = Q @ K^T * rscale --------------------------------------
      for (int t = wv; t < 16; t += WAVES) {
        int mt = t / 4, nt = t % 4;
        int n = nt * 16 + (lane & 15);
        v8f acc = {};
#pragma unroll 1
        for (int kb = 0; kb < 192; kb += 32) {
          v16bf a  = afrag(Q, 192, mt * 16, kb, lane);
          v16bf bb = bfrag(Km + (size_t)n * 192, kb, lane);
          acc = wmma_bf16(a, bb, acc);
        }
#pragma unroll
        for (int v = 0; v < 8; ++v)
          SC[(mt * 16 + v + hiRow) * 64 + n] = acc[v] * rscale;
      }
      __syncthreads();
      // ---- causal + key-pad masked softmax (row per thread, 3 passes) -----
      if (tid < 64) {
        const int vlim = (tid < len - 1) ? tid : (len - 1);  // valid s <= vlim
        float mx = -1e30f;
        for (int s = 0; s <= vlim; ++s) mx = fmaxf(mx, SC[tid * 64 + s]);
        float sum = 0.f;
        for (int s = 0; s <= vlim; ++s) sum += __expf(SC[tid * 64 + s] - mx);
        float inv = 1.f / sum;
        for (int s = 0; s < 64; ++s) {
          float p = (s <= vlim) ? __expf(SC[tid * 64 + s] - mx) * inv : 0.f;
          AT[tid * 64 + s] = (bf16)p;
        }
      }
      __syncthreads();
      // ---- out_h = attn @ V (B-frags contiguous thanks to V^T) ------------
      for (int t = wv; t < 48; t += WAVES) {
        int mt = t / 12, nt = t % 12;
        int n = nt * 16 + (lane & 15);
        v8f acc = {};
#pragma unroll 1
        for (int kb = 0; kb < 64; kb += 32) {
          v16bf a  = afrag(AT, 64, mt * 16, kb, lane);
          v16bf bb = bfrag(VT + (size_t)n * 64, kb, lane);
          acc = wmma_bf16(a, bb, acc);
        }
#pragma unroll
        for (int v = 0; v < 8; ++v)
          OH[(mt * 16 + v + hiRow) * 192 + n] = (bf16)acc[v];
      }
      __syncthreads();
      // ---- x += out_h @ Wo[head rows]  (+ bo once) ------------------------
      for (int t = wv; t < 192; t += WAVES) {
        int mt = t / 48, nt = t % 48;
        int n = nt * 16 + (lane & 15);
        v8f acc = {};
#pragma unroll 1
        for (int kb = 0; kb < 192; kb += 32) {
          v16bf a  = afrag(OH, 192, mt * 16, kb, lane);
          v16bf bb = bfrag(WoT + (size_t)n * 768 + hd * 192, kb, lane);
          acc = wmma_bf16(a, bb, acc);
        }
        float bo = (hd == 0) ? P.bo[l][n] : 0.f;
#pragma unroll
        for (int v = 0; v < 8; ++v) {
          int m = mt * 16 + v + hiRow;
          X[m * 768 + n] = (bf16)((float)X[m * 768 + n] + acc[v] + bo);
        }
      }
      __syncthreads();
    }  // heads

    // ---- FFN: x += gelu(LN2(x) @ W1 + b1) @ W2 + b2 (LN folded in W1T) ----
    row_stats(X, RS, tid);               // ln2 stats
    __syncthreads();
    for (int i = tid; i < 64 * 768; i += BLK) ACC[i] = 0.f;
    __syncthreads();
    for (int kb0 = 0; kb0 < 3072; kb0 += 128) {
      for (int t = wv; t < 32; t += WAVES) {       // I = gelu(h2 @ W1 block)
        int mt = t / 8, nt = t % 8;
        int n = nt * 16 + (lane & 15);
        v8f acc = {};
#pragma unroll 1
        for (int kb = 0; kb < 768; kb += 32) {
          v16bf a  = afrag(X, 768, mt * 16, kb, lane);
          v16bf bb = bfrag(W1T + (size_t)(kb0 + n) * 768, kb, lane);
          acc = wmma_bf16(a, bb, acc);
        }
        float c1v = P.c1[l][kb0 + n], sgv = P.sg1[l][kb0 + n];
#pragma unroll
        for (int v = 0; v < 8; ++v) {
          int m = mt * 16 + v + hiRow;
          float mu = RS[2 * m], rs = RS[2 * m + 1];
          float x = rs * acc[v] + c1v - mu * rs * sgv;
          x = 0.5f * x * (1.f + erff(x * 0.70710678f));   // exact GELU
          IB[m * 128 + n] = (bf16)x;
        }
      }
      __syncthreads();
      for (int t = wv; t < 192; t += WAVES) {      // ACC += I @ W2 block
        int mt = t / 48, nt = t % 48;
        int n = nt * 16 + (lane & 15);
        v8f acc = {};
#pragma unroll 1
        for (int kb = 0; kb < 128; kb += 32) {
          v16bf a  = afrag(IB, 128, mt * 16, kb, lane);
          v16bf bb = bfrag(W2T + (size_t)n * 3072 + kb0, kb, lane);
          acc = wmma_bf16(a, bb, acc);
        }
#pragma unroll
        for (int v = 0; v < 8; ++v)
          ACC[(mt * 16 + v + hiRow) * 768 + n] += acc[v];
      }
      __syncthreads();
    }
    for (int i = tid; i < 64 * 768; i += BLK) {
      int k = i % 768;
      X[i] = (bf16)((float)X[i] + ACC[i] + P.b2[l][k]);
    }
    __syncthreads();
  }  // layers

  // ---- fused pose head: hidden = relu(LN_head(x) @ W1h + b1h) as WMMA -----
  row_stats(X, RS, tid);
  __syncthreads();
  const bf16* HW1T = P.wb + W_HW1T;
  for (int t = wv; t < 32; t += WAVES) {           // 64x128 output, k over 768
    int mt = t / 8, nt = t % 8;
    int n = nt * 16 + (lane & 15);
    v8f acc = {};
#pragma unroll 1
    for (int kb = 0; kb < 768; kb += 32) {
      v16bf a  = afrag(X, 768, mt * 16, kb, lane);
      v16bf bb = bfrag(HW1T + (size_t)n * 768, kb, lane);
      acc = wmma_bf16(a, bb, acc);
    }
    float chv = P.ch[n], sgv = P.sgh[n];
#pragma unroll
    for (int v = 0; v < 8; ++v) {
      int m = mt * 16 + v + hiRow;
      float mu = RS[2 * m], rs = RS[2 * m + 1];
      float x = rs * acc[v] + chv - mu * rs * sgv;
      HIDB[m * 128 + n] = (bf16)fmaxf(x, 0.f);
    }
  }
  __syncthreads();
  // final 128->7 projection; write only the slots pose_seq gathers from here
  for (int idx = tid; idx < 64 * 7; idx += BLK) {
    int slot = idx / 7, o = idx - slot * 7;
    if (slot >= start && slot < start + len) {
      const bf16* hrow = HIDB + (size_t)slot * 128;
      float s = P.hb2[o];
      for (int k = 0; k < 128; ++k) s += (float)hrow[k] * P.hW2[k * 7 + o];
      P.out[((size_t)(b * 64) + slot) * 7 + o] = s;
    }
  }
}

// ---------------------------------------------------------------------------
extern "C" void kernel_launch(void* const* d_in, const int* in_sizes, int n_in,
                              void* d_out, int out_size, void* d_ws, size_t ws_size,
                              hipStream_t stream) {
  (void)in_sizes; (void)n_in; (void)out_size; (void)ws_size;
  const float* feature = (const float*)d_in[0];
  const int LH = 55, LP = 61;            // local_head / local_params bases
  bf16*  wb   = (bf16*)d_ws;
  float* aux  = (float*)((char*)d_ws + AUX_BYTE);
  int*   meta = (int*)((char*)d_ws + META_BYTE);
  bf16*  featbf = (bf16*)((char*)d_ws + FEAT_BYTE);

  KParams P;
  P.featbf = featbf;
  for (int l = 0; l < 4; ++l) {
    const float* W1    = (const float*)d_in[LP + 12 * l + 0];
    const float* W2    = (const float*)d_in[LP + 12 * l + 1];
    const float* Wo    = (const float*)d_in[LP + 12 * l + 2];
    const float* Wqkv  = (const float*)d_in[LP + 12 * l + 3];
    const float* b1    = (const float*)d_in[LP + 12 * l + 4];
    const float* bqkv  = (const float*)d_in[LP + 12 * l + 7];
    const float* ln1b  = (const float*)d_in[LP + 12 * l + 8];
    const float* ln1g  = (const float*)d_in[LP + 12 * l + 9];
    const float* ln2b  = (const float*)d_in[LP + 12 * l + 10];
    const float* ln2g  = (const float*)d_in[LP + 12 * l + 11];
    float* cq  = aux + l * AUX_LAYER;
    float* sgq = cq + 2304;
    float* c1  = sgq + 2304;
    float* sg1 = c1 + 3072;
    fold_ln_transpose<<<(2304 + BLK - 1) / BLK, BLK, 0, stream>>>(
        Wqkv, ln1g, ln1b, bqkv, 768, 2304, wb + (size_t)l * LSTR + W_QKVT, cq, sgq);
    fold_ln_transpose<<<(3072 + BLK - 1) / BLK, BLK, 0, stream>>>(
        W1, ln2g, ln2b, b1, 768, 3072, wb + (size_t)l * LSTR + W_W1T, c1, sg1);
    convert_transpose<<<1024, BLK, 0, stream>>>(Wo, wb + (size_t)l * LSTR + W_WOT, 768, 768);
    convert_transpose<<<2048, BLK, 0, stream>>>(W2, wb + (size_t)l * LSTR + W_W2T, 3072, 768);
    P.bo[l]  = (const float*)d_in[LP + 12 * l + 6];
    P.b2[l]  = (const float*)d_in[LP + 12 * l + 5];
    P.cq[l] = cq; P.sgq[l] = sgq; P.c1[l] = c1; P.sg1[l] = sg1;
  }
  {
    const float* hW1 = (const float*)d_in[LH + 0];
    const float* hb1 = (const float*)d_in[LH + 2];
    const float* hlb = (const float*)d_in[LH + 4];
    const float* hlg = (const float*)d_in[LH + 5];
    float* ch  = aux + AUX_HEAD;
    float* sgh = ch + 128;
    fold_ln_transpose<<<1, BLK, 0, stream>>>(hW1, hlg, hlb, hb1, 768, 128,
                                             wb + W_HW1T, ch, sgh);
    P.ch = ch; P.sgh = sgh;
  }
  convert_cast<<<(FEAT_N + BLK - 1) / BLK, BLK, 0, stream>>>(feature, featbf, FEAT_N);
  scan_chunks<<<1, BLK, 0, stream>>>(feature, meta);

  P.hW2 = (const float*)d_in[LH + 1];
  P.hb2 = (const float*)d_in[LH + 3];
  P.wb = wb; P.meta = meta; P.out = (float*)d_out;

  dim3 grid(64, 8, 1);                   // (chunk index, batch)
  chunk_transformer_pose<<<grid, BLK, SMEM_C, stream>>>(P);
}
